// Attention_4346506903982
// MI455X (gfx1250) — matrix operand order repaired and checked by reading, not compiled, not run on hardware
//
#include <hip/hip_runtime.h>
#include <hip/hip_bf16.h>

// ---------------------------------------------------------------------------
// GQA attention layer for MI455X (gfx1250, wave32, WMMA).
//   ~172 GFLOP, ~250MB HBM traffic (f16 intermediates) -> matrix-op bound.
//   v_wmma_f32_16x16x32_f16 for all GEMM stages; flash attention so the
//   536MB logits tensor never exists; GLOBAL_LOAD_ASYNC_TO_LDS_B128 for the
//   copy-only staging paths; B-operand tiles staged TRANSPOSED in LDS so all
//   WMMA fragment reads are contiguous ds_load_b128 (no scalar u16 gathers).
// ---------------------------------------------------------------------------

typedef __attribute__((ext_vector_type(16))) _Float16 v16h;
typedef __attribute__((ext_vector_type(8)))  float    v8f;
typedef int v4i __attribute__((vector_size(16)));

#define B_   2
#define T_   2048
#define D_   2048
#define NH   16
#define NKV  8
#define HD   128
#define SOFTCAP_F 50.0f
#define BIG_NEG_F -2.3819763e38f

#if defined(__gfx1250__) && __has_builtin(__builtin_amdgcn_global_load_async_to_lds_b128)
#define USE_ASYNC_LDS 1
#else
#define USE_ASYNC_LDS 0
#endif

// 16-byte global->LDS copy: async DMA path on gfx1250, vector copy fallback.
__device__ __forceinline__ void copy16_g2l(const _Float16* gsrc, _Float16* ldst) {
#if USE_ASYNC_LDS
  __builtin_amdgcn_global_load_async_to_lds_b128(
      (__attribute__((address_space(1))) v4i*)gsrc,
      (__attribute__((address_space(3))) v4i*)ldst, 0, 0);
#else
  *reinterpret_cast<float4*>(ldst) = *reinterpret_cast<const float4*>(gsrc);
#endif
}

__device__ __forceinline__ void async_wait() {
#if USE_ASYNC_LDS
#if __has_builtin(__builtin_amdgcn_s_wait_asynccnt)
  __builtin_amdgcn_s_wait_asynccnt(0);
#else
  asm volatile("s_wait_asynccnt 0x0" ::: "memory");
#endif
#endif
}

__device__ __forceinline__ v8f wmma32(v16h a, v16h b, v8f c) {
  // D = A(16x32 f16) * B(32x16 f16) + C(16x16 f32)
  return __builtin_amdgcn_wmma_f32_16x16x32_f16(
      /*neg_a=*/false, a, /*neg_b=*/false, b,
      /*c_mod=*/(short)0, c, /*reuse_a=*/false, /*reuse_b=*/false);
}

// A-matrix fragment: 16x32 f16 tile from row-major [row][k] LDS source.
// k order used here (contiguous; B matches): lane<16 -> rows 0..15, K=0..15; lane>=16 -> K=16..31.
// Contiguous 8-half runs -> vectorizes to ds_load_b128.
__device__ __forceinline__ v16h load_a(const _Float16* p, int ld) {
  const int lane = threadIdx.x & 31;
  const int m = lane & 15, hi = lane >> 4;
  v16h a;
#pragma unroll
  for (int i = 0; i < 16; ++i) {
    const int k = i + 16 * hi;
    a[i] = p[m * ld + k];
  }
  return a;
}

// B-matrix fragment: 32x16 (KxN) from an [n][k] (transposed / n-major) source.
// Each lane reads 16 contiguous halves -> vectorizes to 2x ds_load_b128.
// k order used here (contiguous, same as load_a): lanes 0-15 hold K=0..15, lanes 16-31 hold K=16..31; N = lane&15.
__device__ __forceinline__ v16h load_b_nk(const _Float16* p, int ld) {
  const int lane = threadIdx.x & 31;
  const int n = lane & 15, hi = lane >> 4;
  v16h b;
#pragma unroll
  for (int i = 0; i < 16; ++i) {
    const int k = i + 16 * hi;
    b[i] = p[n * ld + k];
  }
  return b;
}

// Virtual fused weight matrix [D, 4096] = [Wq(2048) | Wk(1024) | Wv(1024)].
__device__ __forceinline__ float w_fetch(const float* __restrict__ wq,
                                         const float* __restrict__ wkv,
                                         int d, int c) {
  if (c < 2048) {                       // Q: w_q[n, d, h], n=c>>7, h=c&127
    return wq[(((size_t)(c >> 7)) * D_ + d) * HD + (c & 127)];
  } else if (c < 3072) {                // K: w_kv[0, n, d, h]
    const int cc = c - 2048;
    return wkv[(((size_t)(cc >> 7)) * D_ + d) * HD + (cc & 127)];
  } else {                              // V: w_kv[1, n, d, h]
    const int cc = c - 3072;
    return wkv[(((size_t)(NKV + (cc >> 7))) * D_ + d) * HD + (cc & 127)];
  }
}

// ---------------------------------------------------------------------------
// Kernel 1: fused QKV projection.  X[4096,2048] x W[2048,4096] -> Q,K,V (f16)
// Block: 128(M) x 64(N) tile, K-steps of 32, 8 waves (4x2), 32x32 per wave.
// ---------------------------------------------------------------------------
__global__ void __launch_bounds__(256) qkv_proj_kernel(
    const float* __restrict__ x, const float* __restrict__ wq,
    const float* __restrict__ wkv, _Float16* __restrict__ Q,
    _Float16* __restrict__ K, _Float16* __restrict__ V) {
  const int bm = blockIdx.y, bn = blockIdx.x;
  const int tid = threadIdx.x;
  const int wave = tid >> 5, lane = tid & 31;
  const int wm = wave >> 1, wn = wave & 1;
  const int hi = lane >> 4, ln = lane & 15;
  const int row0 = bm * 128, col0 = bn * 64;

  __shared__ _Float16 As[128][40];   // 128x32 A tile, row-major
  __shared__ _Float16 Bst[64][40];   // 32x64 B tile stored TRANSPOSED [n][k]

  const v8f vzero = {};
  v8f acc[2][2];
#pragma unroll
  for (int i = 0; i < 2; ++i)
#pragma unroll
    for (int j = 0; j < 2; ++j) acc[i][j] = vzero;

  for (int k0 = 0; k0 < D_; k0 += 32) {
    // stage A: 128x32 f32 -> f16 (float4 per thread x4)
#pragma unroll
    for (int i = 0; i < 4; ++i) {
      const int e = (tid + i * 256) * 4;
      const int r = e >> 5, c = e & 31;
      const float4 xv =
          *reinterpret_cast<const float4*>(&x[(size_t)(row0 + r) * D_ + k0 + c]);
      As[r][c + 0] = (_Float16)xv.x; As[r][c + 1] = (_Float16)xv.y;
      As[r][c + 2] = (_Float16)xv.z; As[r][c + 3] = (_Float16)xv.w;
    }
    // stage B transposed: element (k, c) -> Bst[c][k]  (scalar gather anyway)
#pragma unroll
    for (int i = 0; i < 8; ++i) {
      const int e = tid + i * 256;
      const int k = e >> 6, c = e & 63;
      Bst[c][k] = (_Float16)w_fetch(wq, wkv, k0 + k, col0 + c);
    }
    __syncthreads();

    v16h af[2], bf[2];
#pragma unroll
    for (int ti = 0; ti < 2; ++ti) af[ti] = load_a(&As[wm * 32 + ti * 16][0], 40);
#pragma unroll
    for (int tj = 0; tj < 2; ++tj) bf[tj] = load_b_nk(&Bst[wn * 32 + tj * 16][0], 40);
#pragma unroll
    for (int ti = 0; ti < 2; ++ti)
#pragma unroll
      for (int tj = 0; tj < 2; ++tj) acc[ti][tj] = wmma32(af[ti], bf[tj], acc[ti][tj]);
    __syncthreads();
  }

  // Routing is uniform per block (64-wide tile never straddles Q/K/V regions).
  _Float16* dst;
  int ldd, cbase;
  if (col0 < 2048)      { dst = Q; ldd = 2048; cbase = col0; }
  else if (col0 < 3072) { dst = K; ldd = 1024; cbase = col0 - 2048; }
  else                  { dst = V; ldd = 1024; cbase = col0 - 3072; }

#pragma unroll
  for (int ti = 0; ti < 2; ++ti)
#pragma unroll
    for (int tj = 0; tj < 2; ++tj)
#pragma unroll
      for (int j = 0; j < 8; ++j) {
        const int gr = row0 + wm * 32 + ti * 16 + j + 8 * hi;
        const int gc = cbase + wn * 32 + tj * 16 + ln;
        dst[(size_t)gr * ldd + gc] = (_Float16)acc[ti][tj][j];
      }
}

// ---------------------------------------------------------------------------
// Kernel 2: RoPE on Q (with 1/sqrt(HD) scale) and K, in place on f16.
// ---------------------------------------------------------------------------
__global__ void __launch_bounds__(256) rope_kernel(
    _Float16* __restrict__ Q, _Float16* __restrict__ K,
    const int* __restrict__ pos) {
  const int idx = blockIdx.x * 256 + threadIdx.x;
  const int total = B_ * T_ * (NH + NKV) * (HD / 2);
  if (idx >= total) return;
  const int i = idx & 63;
  int rest = idx >> 6;
  const int h24 = rest % (NH + NKV); rest /= (NH + NKV);
  const int t = rest % T_;
  const int b = rest / T_;

  const float p = (float)pos[b * T_ + t];
  // angle = p * ROPE_BASE^(-i/64);  -ln(10000)/64 = -0.14391156831212787
  const float ang = p * __expf((float)i * -0.14391156831212787f);
  float s, c;
  __sincosf(ang, &s, &c);

  _Float16* base;
  float scale = 1.0f;
  if (h24 < NH) {
    base = Q + ((size_t)(b * T_ + t) * NH + h24) * HD;
    scale = 0.08838834764831845f;  // HD^-0.5
  } else {
    base = K + ((size_t)(b * T_ + t) * NKV + (h24 - NH)) * HD;
  }
  const float x1 = (float)base[i], x2 = (float)base[i + 64];
  base[i]      = (_Float16)((x1 * c - x2 * s) * scale);
  base[i + 64] = (_Float16)((x2 * c + x1 * s) * scale);
}

// ---------------------------------------------------------------------------
// Kernel 3: flash attention (causal, tanh softcap) for one (b, head, Qtile).
// Q/K tiles async-DMA'd to LDS; V staged transposed [h][s] so PV B-fragments
// are contiguous.  8 waves: S(64x32)=1 tile/wave, O(64x128)=4 tiles/wave.
// ---------------------------------------------------------------------------
__global__ void __launch_bounds__(256) attn_kernel(
    const _Float16* __restrict__ Q, const _Float16* __restrict__ K,
    const _Float16* __restrict__ V, _Float16* __restrict__ E) {
  const int qi = blockIdx.x;        // 0..31 (64-row q tiles)
  const int qh = blockIdx.y;        // 0..15 q head; kv head = qh>>1 (G=2)
  const int b  = blockIdx.z;
  const int kh = qh >> 1;
  const int tid = threadIdx.x;
  const int wave = tid >> 5, lane = tid & 31;
  const int hi = lane >> 4, ln = lane & 15;
  const int qbase = qi * 64;

  __shared__ _Float16 Qs[64][136];   // Q tile [q][h]
  __shared__ _Float16 Ks[32][136];   // K tile [s][h]  (= [n][k] for QK^T)
  __shared__ _Float16 Vst[128][40];  // V tile TRANSPOSED [h][s]
  __shared__ float    Ss[64][36];
  __shared__ _Float16 Ps[64][40];
  __shared__ float rowM[64], rowL[64], rowA[64];

  // stage Q tile: pure f16 copy -> async DMA (16 rows per pass)
#pragma unroll
  for (int i = 0; i < 4; ++i) {
    const int e = (tid + i * 256) * 8;          // halves
    const int r = e >> 7, c = e & 127;
    copy16_g2l(&Q[((size_t)(b * T_ + qbase + r) * NH + qh) * HD + c], &Qs[r][c]);
  }
  if (tid < 64) { rowM[tid] = BIG_NEG_F; rowL[tid] = 0.0f; }

  const int rm = wave >> 1;          // O row tile (16 rows)
  const int cb = (wave & 1) * 64;    // O col base (4x16 cols)
  const v8f vzero = {};
  v8f oacc[4];
#pragma unroll
  for (int t = 0; t < 4; ++t) oacc[t] = vzero;

  const int ntiles = qi * 2 + 2;     // causal: keys up to qbase+63
  for (int it = 0; it < ntiles; ++it) {
    const int s0 = it * 32;
    __syncthreads();  // previous iteration finished with Ks/Vst
#pragma unroll
    for (int i = 0; i < 2; ++i) {
      const int e = (tid + i * 256) * 8;        // halves
      const int r = e >> 7, c = e & 127;
      const size_t kvbase = ((size_t)(b * T_ + s0 + r) * NKV + kh) * HD + c;
      // K: straight copy -> async DMA
      copy16_g2l(&K[kvbase], &Ks[r][c]);
      // V: transpose into Vst[h][s] (scalar scatter; makes PV reads b128)
      const float4 vv = *reinterpret_cast<const float4*>(&V[kvbase]);
      const _Float16* hp = reinterpret_cast<const _Float16*>(&vv);
#pragma unroll
      for (int u = 0; u < 8; ++u) Vst[c + u][r] = hp[u];
    }
    async_wait();
    __syncthreads();

    // ---- S = Q K^T (64x32), softcap + causal mask, into LDS ----
    {
      const int srm = wave >> 1;     // 0..3
      const int scn = wave & 1;      // 0..1
      v8f s = vzero;
#pragma unroll
      for (int ks = 0; ks < 4; ++ks) {   // contraction over HD=128
        v16h a  = load_a(&Qs[srm * 16][ks * 32], 136);
        v16h bk = load_b_nk(&Ks[scn * 16][ks * 32], 136);
        s = wmma32(a, bk, s);
      }
#pragma unroll
      for (int j = 0; j < 8; ++j) {
        const int r  = srm * 16 + j + 8 * hi;
        const int cc = scn * 16 + ln;
        float sv = tanhf(s[j] * (1.0f / SOFTCAP_F)) * SOFTCAP_F;
        if (s0 + cc > qbase + r) sv = BIG_NEG_F;   // causal
        Ss[r][cc] = sv;
      }
    }
    __syncthreads();

    // ---- online softmax: one thread per row ----
    if (tid < 64) {
      const int r = tid;
      const float mold = rowM[r];
      float mx = mold;
#pragma unroll
      for (int c = 0; c < 32; ++c) mx = fmaxf(mx, Ss[r][c]);
      const float alpha = __expf(mold - mx);
      float sum = 0.0f;
#pragma unroll
      for (int c = 0; c < 32; ++c) {
        const float pv = __expf(Ss[r][c] - mx);
        Ps[r][c] = (_Float16)pv;
        sum += pv;
      }
      rowA[r] = alpha;
      rowM[r] = mx;
      rowL[r] = rowL[r] * alpha + sum;
    }
    __syncthreads();

    // ---- rescale O, then O += P * V (K-dim 32 = one WMMA step) ----
#pragma unroll
    for (int tt = 0; tt < 4; ++tt)
#pragma unroll
      for (int j = 0; j < 8; ++j)
        oacc[tt][j] *= rowA[rm * 16 + j + 8 * hi];

    const v16h ap = load_a(&Ps[rm * 16][0], 40);
#pragma unroll
    for (int tt = 0; tt < 4; ++tt) {
      v16h bv = load_b_nk(&Vst[cb + tt * 16][0], 40);   // B[k][n] = Vst[col+n][k]
      oacc[tt] = wmma32(ap, bv, oacc[tt]);
    }
  }
  __syncthreads();

  // normalize and emit encoded [B,T,NH,HD] (f16)
#pragma unroll
  for (int tt = 0; tt < 4; ++tt)
#pragma unroll
    for (int j = 0; j < 8; ++j) {
      const int r = rm * 16 + j + 8 * hi;
      const int c = cb + tt * 16 + ln;
      E[((size_t)(b * T_ + qbase + r) * NH + qh) * HD + c] =
          (_Float16)(oacc[tt][j] / rowL[r]);
    }
}

// ---------------------------------------------------------------------------
// Kernel 4: output projection.  E[4096,2048](f16) x Wo[2048,2048](f32) -> out.
// (w_out[n,h,d] flattens to row-major [n*128+h][d], matching E's layout.)
// ---------------------------------------------------------------------------
__global__ void __launch_bounds__(256) out_proj_kernel(
    const _Float16* __restrict__ E, const float* __restrict__ Wo,
    float* __restrict__ out) {
  const int bm = blockIdx.y, bn = blockIdx.x;
  const int tid = threadIdx.x;
  const int wave = tid >> 5, lane = tid & 31;
  const int wm = wave >> 1, wn = wave & 1;
  const int hi = lane >> 4, ln = lane & 15;
  const int row0 = bm * 128, col0 = bn * 64;

  __shared__ _Float16 As[128][40];   // 128x32 A tile, row-major
  __shared__ _Float16 Bst[64][40];   // 32x64 B tile stored TRANSPOSED [n][k]

  const v8f vzero = {};
  v8f acc[2][2];
#pragma unroll
  for (int i = 0; i < 2; ++i)
#pragma unroll
    for (int j = 0; j < 2; ++j) acc[i][j] = vzero;

  for (int k0 = 0; k0 < 2048; k0 += 32) {
    // stage A: 128x32 f16 copy -> async DMA
#pragma unroll
    for (int i = 0; i < 2; ++i) {
      const int e = (tid + i * 256) * 8;        // halves
      const int r = e >> 5, c = e & 31;
      copy16_g2l(&E[(size_t)(row0 + r) * 2048 + k0 + c], &As[r][c]);
    }
    // stage B transposed: Wo element (k, c) -> Bst[c][k]
#pragma unroll
    for (int i = 0; i < 2; ++i) {
      const int e = (tid + i * 256) * 4;
      const int k = e >> 6, c = e & 63;
      const float4 wv =
          *reinterpret_cast<const float4*>(&Wo[(size_t)(k0 + k) * D_ + col0 + c]);
      Bst[c + 0][k] = (_Float16)wv.x; Bst[c + 1][k] = (_Float16)wv.y;
      Bst[c + 2][k] = (_Float16)wv.z; Bst[c + 3][k] = (_Float16)wv.w;
    }
    async_wait();
    __syncthreads();

    v16h af[2], bf[2];
#pragma unroll
    for (int ti = 0; ti < 2; ++ti) af[ti] = load_a(&As[wm * 32 + ti * 16][0], 40);
#pragma unroll
    for (int tj = 0; tj < 2; ++tj) bf[tj] = load_b_nk(&Bst[wn * 32 + tj * 16][0], 40);
#pragma unroll
    for (int ti = 0; ti < 2; ++ti)
#pragma unroll
      for (int tj = 0; tj < 2; ++tj) acc[ti][tj] = wmma32(af[ti], bf[tj], acc[ti][tj]);
    __syncthreads();
  }

#pragma unroll
  for (int ti = 0; ti < 2; ++ti)
#pragma unroll
    for (int tj = 0; tj < 2; ++tj)
#pragma unroll
      for (int j = 0; j < 8; ++j) {
        const int gr = row0 + wm * 32 + ti * 16 + j + 8 * hi;
        const int gc = col0 + wn * 32 + tj * 16 + ln;
        out[(size_t)gr * D_ + gc] = acc[ti][tj][j];
      }
}

// ---------------------------------------------------------------------------
// Launcher. Workspace (f16): Q[4096x2048] K[4096x1024] V[4096x1024] E[4096x2048]
// ---------------------------------------------------------------------------
extern "C" void kernel_launch(void* const* d_in, const int* in_sizes, int n_in,
                              void* d_out, int out_size, void* d_ws, size_t ws_size,
                              hipStream_t stream) {
  const float* x     = (const float*)d_in[0];
  const int*   pos   = (const int*)d_in[1];
  // d_in[2] = attn_mask (causal; computed analytically, not read)
  const float* w_q   = (const float*)d_in[3];
  const float* w_kv  = (const float*)d_in[4];
  const float* w_out = (const float*)d_in[5];
  float* out = (float*)d_out;

  _Float16* ws = (_Float16*)d_ws;
  _Float16* Qh = ws;                       // 8388608 halves
  _Float16* Kh = ws + 8388608;             // 4194304
  _Float16* Vh = ws + 12582912;            // 4194304
  _Float16* Eh = ws + 16777216;            // 8388608

  qkv_proj_kernel<<<dim3(64, 32), 256, 0, stream>>>(x, w_q, w_kv, Qh, Kh, Vh);

  const int rope_total = B_ * T_ * (NH + NKV) * (HD / 2);
  rope_kernel<<<(rope_total + 255) / 256, 256, 0, stream>>>(Qh, Kh, pos);

  attn_kernel<<<dim3(T_ / 64, NH, B_), 256, 0, stream>>>(Qh, Kh, Vh, Eh);

  out_proj_kernel<<<dim3(D_ / 64, (B_ * T_) / 128), 256, 0, stream>>>(Eh, w_out, out);
}
